// Qwen3_5GatedDeltaNet_59768764891943
// MI455X (gfx1250) — matrix operand order repaired and checked by reading, not compiled, not run on hardware
//
#include <hip/hip_runtime.h>

// ---------------- problem constants ----------------
#define T_LEN    4096
#define HID      2048
#define HK       16
#define HV       32
#define DK       128
#define DV       128
#define KEY_DIM  (HK * DK)                 // 2048
#define VAL_DIM  (HV * DV)                 // 4096
#define CONV_DIM (2 * KEY_DIM + VAL_DIM)   // 8192
#define EPS      1e-6f

// ---------------- WMMA types ----------------
typedef __attribute__((ext_vector_type(16))) __bf16         v16bf;
typedef __attribute__((ext_vector_type(8)))  float          v8f;
typedef __attribute__((ext_vector_type(8)))  unsigned short u16x8;

union Frag {
  uint4 u[2];   // 32 bytes = 16 bf16
  v16bf v;
};

__device__ __forceinline__ unsigned short f2bf(float f) {
  unsigned int u = __float_as_uint(f);
  unsigned int r = u + 0x7FFFu + ((u >> 16) & 1u);   // round-to-nearest-even
  return (unsigned short)(r >> 16);
}

__device__ __forceinline__ float dot4(const float4& a, const float4& b) {
  return fmaf(a.x, b.x, fmaf(a.y, b.y, fmaf(a.z, b.z, a.w * b.w)));
}

// ---------------- fp32 -> bf16 conversion ----------------
__global__ void cvt_bf16_kernel(const float* __restrict__ src,
                                unsigned short* __restrict__ dst, size_t n) {
  size_t i = (size_t)blockIdx.x * blockDim.x + threadIdx.x;
  size_t stride = (size_t)gridDim.x * blockDim.x;
  for (; i < n; i += stride) dst[i] = f2bf(src[i]);
}

// ---------------- bf16 WMMA GEMM: C[M,N] = A[M,K] * B[K,N] ----------------
// A row-major bf16, B row-major bf16, C fp32. M%128==0, K%32==0.
// NGUARD=false requires N%128==0 (no bounds checks anywhere).
#define BM 128
#define BN 128
#define BK 32

template <bool NGUARD>
__global__ __launch_bounds__(256) void wmma_gemm_bf16(
    const unsigned short* __restrict__ A,
    const unsigned short* __restrict__ B,
    float* __restrict__ C, int M, int N, int K) {
  __shared__ __align__(16) unsigned short As[2][BM * BK];  // [m][k] row-major
  __shared__ __align__(16) unsigned short Bs[2][BN * BK];  // [n][k] transposed

  const int tid  = threadIdx.x;
  const int lane = tid & 31;
  const int wid  = tid >> 5;   // 0..7
  const int wm   = wid & 3;    // wave M strip (32 rows)
  const int wn   = wid >> 2;   // wave N strip (64 cols)
  const int bm   = blockIdx.y * BM;
  const int bn   = blockIdx.x * BN;
  const int hl   = lane & 15;
  const int hs   = lane >> 4;

  // A staging: 512 slots of (row, 8-col group), 2 per thread
  const int arow0 = tid >> 2,          acg0 = tid & 3;
  const int arow1 = (tid + 256) >> 2,  acg1 = (tid + 256) & 3;
  // B staging: each thread owns 2 adjacent K-rows x 8 columns, so the
  // transposed stash packs {B[kr][n], B[kr+1][n]} into one dword store.
  const int bkr = (tid >> 4) * 2;      // 0,2,...,30
  const int bng = tid & 15;            // 8-column group

  v8f acc[2][4];
#pragma unroll
  for (int i = 0; i < 2; ++i)
#pragma unroll
    for (int j = 0; j < 4; ++j) acc[i][j] = {};

  auto loadA = [&](int k0, uint4& a0, uint4& a1) {
    a0 = *(const uint4*)(A + (size_t)(bm + arow0) * K + k0 + acg0 * 8);
    a1 = *(const uint4*)(A + (size_t)(bm + arow1) * K + k0 + acg1 * 8);
  };
  auto loadBrow = [&](int krow) -> u16x8 {
    int nb = bn + bng * 8;
    u16x8 d;
    if (!NGUARD || nb + 8 <= N) {
      d = *(const u16x8*)(B + (size_t)krow * N + nb);
    } else {
#pragma unroll
      for (int j = 0; j < 8; ++j)
        d[j] = (nb + j < N) ? B[(size_t)krow * N + nb + j] : (unsigned short)0;
    }
    return d;
  };
  auto stash = [&](int st, const uint4& a0, const uint4& a1, const u16x8& blo,
                   const u16x8& bhi) {
    *(uint4*)&As[st][arow0 * BK + acg0 * 8] = a0;
    *(uint4*)&As[st][arow1 * BK + acg1 * 8] = a1;
#pragma unroll
    for (int j = 0; j < 8; ++j) {
      unsigned int pk = (unsigned int)blo[j] | ((unsigned int)bhi[j] << 16);
      *(unsigned int*)&Bs[st][(bng * 8 + j) * BK + bkr] = pk;  // 4B aligned
    }
  };

  uint4 a0, a1;
  u16x8 blo, bhi;
  loadA(0, a0, a1);
  blo = loadBrow(bkr);
  bhi = loadBrow(bkr + 1);
  stash(0, a0, a1, blo, bhi);
  __syncthreads();

  const int niter = K / BK;
  for (int i = 0; i < niter; ++i) {
    const int p = i & 1;
    const bool more = (i + 1 < niter);
    if (more) {
      // issue next tile's global loads; latency hides under the WMMAs below
      const int kn0 = (i + 1) * BK;
      loadA(kn0, a0, a1);
      blo = loadBrow(kn0 + bkr);
      bhi = loadBrow(kn0 + bkr + 1);
      if (i + 2 < niter) {
        __builtin_prefetch(A + (size_t)(bm + arow0) * K + (i + 2) * BK, 0, 1);
        __builtin_prefetch(B + (size_t)((i + 2) * BK + bkr) * N + bn, 0, 1);
      }
    }

    // A fragment per ISA 16-bit A 16x32 layout:
    //   lanes 0-15: row=m, K {0..7,16..23}; lanes 16-31: row=m, K {8..15,24..31}
    Frag af[2];
#pragma unroll
    for (int sm = 0; sm < 2; ++sm) {
      int r  = wm * 32 + sm * 16 + hl;
      int kf = hs * 8;
      af[sm].u[0] = *(const uint4*)&As[p][r * BK + kf];
      af[sm].u[1] = *(const uint4*)&As[p][r * BK + kf + 16];
    }
    // B fragment: lanes 0-15 hold K {0..7,16..23} of col n; lanes 16-31 hold K {8..15,24..31}
    Frag bfr[4];
#pragma unroll
    for (int sn = 0; sn < 4; ++sn) {
      int n  = wn * 64 + sn * 16 + hl;
      int kf = hs * 8;
      bfr[sn].u[0] = *(const uint4*)&Bs[p][n * BK + kf];
      bfr[sn].u[1] = *(const uint4*)&Bs[p][n * BK + kf + 16];
    }
#pragma unroll
    for (int sm = 0; sm < 2; ++sm)
#pragma unroll
      for (int sn = 0; sn < 4; ++sn)
        acc[sm][sn] = __builtin_amdgcn_wmma_f32_16x16x32_bf16(
            false, af[sm].v, false, bfr[sn].v, (short)0, acc[sm][sn],
            false, false);

    if (more) stash(p ^ 1, a0, a1, blo, bhi);
    __syncthreads();
  }

  // C/D f32 layout: VGPR r -> row r (lanes 0-15) / row 8+r (lanes 16-31)
#pragma unroll
  for (int sm = 0; sm < 2; ++sm) {
    int rbase = bm + wm * 32 + sm * 16 + hs * 8;
#pragma unroll
    for (int sn = 0; sn < 4; ++sn) {
      int col = bn + wn * 64 + sn * 16 + hl;
      if (!NGUARD || col < N) {
#pragma unroll
        for (int r = 0; r < 8; ++r)
          C[(size_t)(rbase + r) * N + col] = acc[sm][sn][r];
      }
    }
  }
}

// ---------------- causal depthwise conv (K=4) + silu ----------------
__global__ __launch_bounds__(256) void conv_silu_kernel(
    const float* __restrict__ mixed, const float* __restrict__ conv_w,
    float* __restrict__ out) {
  size_t idx = (size_t)blockIdx.x * blockDim.x + threadIdx.x;
  int t = (int)(idx / CONV_DIM);
  int c = (int)(idx % CONV_DIM);
  float4 w = *(const float4*)(conv_w + (size_t)c * 4);
  float acc = mixed[idx] * w.w;                                    // j=3
  if (t >= 1) acc = fmaf(mixed[idx - CONV_DIM], w.z, acc);         // j=2
  if (t >= 2) acc = fmaf(mixed[idx - 2 * (size_t)CONV_DIM], w.y, acc);
  if (t >= 3) acc = fmaf(mixed[idx - 3 * (size_t)CONV_DIM], w.x, acc);
  out[idx] = acc / (1.0f + __expf(-acc));                          // silu
}

// ---------------- per-head L2 norm of q,k (one wave per head) ----------------
__global__ __launch_bounds__(512) void qknorm_kernel(
    const float* __restrict__ convo, float* __restrict__ qn,
    float* __restrict__ kn) {
  const int t = blockIdx.x;
  const int w = threadIdx.x >> 5;     // head 0..15
  const int lane = threadIdx.x & 31;
  {
    const float* src = convo + (size_t)t * CONV_DIM + w * DK + lane * 4;
    float4 x = *(const float4*)src;
    float ss = dot4(x, x);
#pragma unroll
    for (int m = 16; m >= 1; m >>= 1) ss += __shfl_xor(ss, m);
    float rn = rsqrtf(ss + EPS) * 0.0883883476483184f;  // * DK^-0.5
    float4 y = {x.x * rn, x.y * rn, x.z * rn, x.w * rn};
    *(float4*)(qn + ((size_t)t * HK + w) * DK + lane * 4) = y;
  }
  {
    const float* src =
        convo + (size_t)t * CONV_DIM + KEY_DIM + w * DK + lane * 4;
    float4 x = *(const float4*)src;
    float ss = dot4(x, x);
#pragma unroll
    for (int m = 16; m >= 1; m >>= 1) ss += __shfl_xor(ss, m);
    float rn = rsqrtf(ss + EPS);
    float4 y = {x.x * rn, x.y * rn, x.z * rn, x.w * rn};
    *(float4*)(kn + ((size_t)t * HK + w) * DK + lane * 4) = y;
  }
}

// ---------------- beta = sigmoid(b), g = -exp(A_log)*softplus_t(a+dt_bias) ----
__global__ void gates_kernel(const float* __restrict__ b,
                             const float* __restrict__ a,
                             const float* __restrict__ A_log,
                             const float* __restrict__ dt_bias,
                             float* __restrict__ g, float* __restrict__ beta) {
  int idx = blockIdx.x * blockDim.x + threadIdx.x;
  if (idx >= T_LEN * HV) return;
  int h = idx & (HV - 1);
  beta[idx] = 1.0f / (1.0f + __expf(-b[idx]));
  float sp = a[idx] + dt_bias[h];
  sp = (sp > 20.0f) ? sp : log1pf(__expf(sp));
  g[idx] = -__expf(A_log[h]) * sp;
}

// ---------------- gated delta rule recurrence ----------------
// Columns of the per-head state evolve independently. One wave per 4 DV
// columns; lane l holds S[4l..4l+3] for each column in registers. grid=HV*4.
__global__ __launch_bounds__(256) void recur_kernel(
    const float* __restrict__ qn, const float* __restrict__ kn,
    const float* __restrict__ convo, const float* __restrict__ g,
    const float* __restrict__ beta, float* __restrict__ o) {
  const int lane = threadIdx.x & 31;
  const int wave = threadIdx.x >> 5;              // 0..7
  const int h    = blockIdx.x >> 2;               // 0..31
  const int dvb  = (blockIdx.x & 3) * 32 + wave * 4;
  const int kh   = h >> 1;                        // GQA rep = 2

  float4 s0 = {0, 0, 0, 0}, s1 = {0, 0, 0, 0}, s2 = {0, 0, 0, 0},
         s3 = {0, 0, 0, 0};

  const float* vptr  = convo + 2 * KEY_DIM + h * DV + dvb;
  const float* kbase = kn + (size_t)kh * DK + lane * 4;
  const float* qbase = qn + (size_t)kh * DK + lane * 4;

  for (int t = 0; t < T_LEN; ++t) {
    float gt = g[t * HV + h];
    float bt = beta[t * HV + h];
    float eg = __expf(gt);
    float4 k4 = *(const float4*)(kbase + (size_t)t * (HK * DK));
    float4 q4 = *(const float4*)(qbase + (size_t)t * (HK * DK));
    float4 v4 = *(const float4*)(vptr + (size_t)t * CONV_DIM);

    float d0 = dot4(k4, s0), d1 = dot4(k4, s1), d2 = dot4(k4, s2),
          d3 = dot4(k4, s3);
#pragma unroll
    for (int m = 16; m >= 1; m >>= 1) {
      d0 += __shfl_xor(d0, m);
      d1 += __shfl_xor(d1, m);
      d2 += __shfl_xor(d2, m);
      d3 += __shfl_xor(d3, m);
    }
    float de0 = (v4.x - eg * d0) * bt;
    float de1 = (v4.y - eg * d1) * bt;
    float de2 = (v4.z - eg * d2) * bt;
    float de3 = (v4.w - eg * d3) * bt;

    s0.x = fmaf(s0.x, eg, k4.x * de0); s0.y = fmaf(s0.y, eg, k4.y * de0);
    s0.z = fmaf(s0.z, eg, k4.z * de0); s0.w = fmaf(s0.w, eg, k4.w * de0);
    s1.x = fmaf(s1.x, eg, k4.x * de1); s1.y = fmaf(s1.y, eg, k4.y * de1);
    s1.z = fmaf(s1.z, eg, k4.z * de1); s1.w = fmaf(s1.w, eg, k4.w * de1);
    s2.x = fmaf(s2.x, eg, k4.x * de2); s2.y = fmaf(s2.y, eg, k4.y * de2);
    s2.z = fmaf(s2.z, eg, k4.z * de2); s2.w = fmaf(s2.w, eg, k4.w * de2);
    s3.x = fmaf(s3.x, eg, k4.x * de3); s3.y = fmaf(s3.y, eg, k4.y * de3);
    s3.z = fmaf(s3.z, eg, k4.z * de3); s3.w = fmaf(s3.w, eg, k4.w * de3);

    float o0 = dot4(q4, s0), o1 = dot4(q4, s1), o2 = dot4(q4, s2),
          o3 = dot4(q4, s3);
#pragma unroll
    for (int m = 16; m >= 1; m >>= 1) {
      o0 += __shfl_xor(o0, m);
      o1 += __shfl_xor(o1, m);
      o2 += __shfl_xor(o2, m);
      o3 += __shfl_xor(o3, m);
    }
    if (lane == 0) {
      float4 ov = {o0, o1, o2, o3};
      *(float4*)(o + ((size_t)t * HV + h) * DV + dvb) = ov;
    }
  }
}

// ---------------- gated RMSNorm + silu(z) gate, output as bf16 ----------------
__global__ __launch_bounds__(256) void gnorm_kernel(
    const float* __restrict__ o, const float* __restrict__ z,
    const float* __restrict__ norm_w, unsigned short* __restrict__ ogb) {
  const int t = blockIdx.x;
  const int w = threadIdx.x >> 5;
  const int lane = threadIdx.x & 31;
  float4 nw = *(const float4*)(norm_w + lane * 4);
#pragma unroll
  for (int hh = 0; hh < 4; ++hh) {
    int h = w + hh * 8;
    size_t base = ((size_t)t * HV + h) * DV + lane * 4;
    float4 x = *(const float4*)(o + base);
    float ss = dot4(x, x);
#pragma unroll
    for (int m = 16; m >= 1; m >>= 1) ss += __shfl_xor(ss, m);
    float r = rsqrtf(ss * (1.0f / DV) + EPS);
    float4 zz = *(const float4*)(z + base);
    ushort4 res;
    res.x = f2bf(x.x * r * nw.x * (zz.x / (1.0f + __expf(-zz.x))));
    res.y = f2bf(x.y * r * nw.y * (zz.y / (1.0f + __expf(-zz.y))));
    res.z = f2bf(x.z * r * nw.z * (zz.z / (1.0f + __expf(-zz.z))));
    res.w = f2bf(x.w * r * nw.w * (zz.w / (1.0f + __expf(-zz.w))));
    *(ushort4*)(ogb + base) = res;
  }
}

// ---------------- launcher ----------------
extern "C" void kernel_launch(void* const* d_in, const int* in_sizes, int n_in,
                              void* d_out, int out_size, void* d_ws,
                              size_t ws_size, hipStream_t stream) {
  (void)in_sizes; (void)n_in; (void)out_size; (void)ws_size;
  const float* x       = (const float*)d_in[0];
  const float* W_qkv   = (const float*)d_in[1];
  const float* W_z     = (const float*)d_in[2];
  const float* W_b     = (const float*)d_in[3];
  const float* W_a     = (const float*)d_in[4];
  const float* conv_w  = (const float*)d_in[5];
  const float* A_log   = (const float*)d_in[6];
  const float* dt_bias = (const float*)d_in[7];
  const float* norm_w  = (const float*)d_in[8];
  const float* W_out   = (const float*)d_in[9];
  float* out = (float*)d_out;

  char* ws = (char*)d_ws;
  size_t off = 0;
  auto alloc = [&](size_t bytes) -> void* {
    void* p = ws + off;
    off = (off + bytes + 255) & ~(size_t)255;
    return p;
  };

  unsigned short* xb    = (unsigned short*)alloc((size_t)T_LEN * HID * 2);
  unsigned short* wqkvb = (unsigned short*)alloc((size_t)HID * CONV_DIM * 2);
  unsigned short* wzb   = (unsigned short*)alloc((size_t)HID * VAL_DIM * 2);
  unsigned short* wbb   = (unsigned short*)alloc((size_t)HID * HV * 2);
  unsigned short* wab   = (unsigned short*)alloc((size_t)HID * HV * 2);
  unsigned short* woutb = (unsigned short*)alloc((size_t)VAL_DIM * HID * 2);
  float* mixed = (float*)alloc((size_t)T_LEN * CONV_DIM * 4);  // reused below
  float* convo = (float*)alloc((size_t)T_LEN * CONV_DIM * 4);  // reused below
  float* zbuf  = (float*)alloc((size_t)T_LEN * VAL_DIM * 4);
  float* bcol  = (float*)alloc((size_t)T_LEN * HV * 4);
  float* acol  = (float*)alloc((size_t)T_LEN * HV * 4);
  float* gbuf  = (float*)alloc((size_t)T_LEN * HV * 4);
  float* betab = (float*)alloc((size_t)T_LEN * HV * 4);
  // aliases: mixed (T*8192 f32) is dead after conv_silu -> holds qn|kn|o
  float* qn   = mixed;                                   // T*HK*DK
  float* kn   = mixed + (size_t)T_LEN * HK * DK;         // T*HK*DK
  float* obuf = mixed + (size_t)2 * T_LEN * HK * DK;     // T*HV*DV (fits exactly)
  // convo is dead after the recurrence -> holds bf16 gated output
  unsigned short* ogb = (unsigned short*)convo;          // T*VAL_DIM bf16

  // fp32 -> bf16 conversions
  cvt_bf16_kernel<<<2048, 256, 0, stream>>>(x, xb, (size_t)T_LEN * HID);
  cvt_bf16_kernel<<<2048, 256, 0, stream>>>(W_qkv, wqkvb, (size_t)HID * CONV_DIM);
  cvt_bf16_kernel<<<2048, 256, 0, stream>>>(W_z, wzb, (size_t)HID * VAL_DIM);
  cvt_bf16_kernel<<<256, 256, 0, stream>>>(W_b, wbb, (size_t)HID * HV);
  cvt_bf16_kernel<<<256, 256, 0, stream>>>(W_a, wab, (size_t)HID * HV);
  cvt_bf16_kernel<<<2048, 256, 0, stream>>>(W_out, woutb, (size_t)VAL_DIM * HID);

  // input projections (WMMA, unguarded fast path: N % 128 == 0)
  {
    dim3 g(CONV_DIM / BN, T_LEN / BM);
    wmma_gemm_bf16<false><<<g, 256, 0, stream>>>(xb, wqkvb, mixed, T_LEN,
                                                 CONV_DIM, HID);
  }
  {
    dim3 g(VAL_DIM / BN, T_LEN / BM);
    wmma_gemm_bf16<false><<<g, 256, 0, stream>>>(xb, wzb, zbuf, T_LEN, VAL_DIM,
                                                 HID);
  }
  // small N=32 projections (guarded instance)
  {
    dim3 g(1, T_LEN / BM);
    wmma_gemm_bf16<true><<<g, 256, 0, stream>>>(xb, wbb, bcol, T_LEN, HV, HID);
    wmma_gemm_bf16<true><<<g, 256, 0, stream>>>(xb, wab, acol, T_LEN, HV, HID);
  }

  conv_silu_kernel<<<(T_LEN * CONV_DIM) / 256, 256, 0, stream>>>(mixed, conv_w,
                                                                 convo);
  qknorm_kernel<<<T_LEN, 512, 0, stream>>>(convo, qn, kn);
  gates_kernel<<<(T_LEN * HV) / 256, 256, 0, stream>>>(bcol, acol, A_log,
                                                       dt_bias, gbuf, betab);
  recur_kernel<<<HV * 4, 256, 0, stream>>>(qn, kn, convo, gbuf, betab, obuf);
  gnorm_kernel<<<T_LEN, 256, 0, stream>>>(obuf, zbuf, norm_w, ogb);

  // output projection (WMMA, unguarded fast path)
  {
    dim3 g(HID / BN, T_LEN / BM);
    wmma_gemm_bf16<false><<<g, 256, 0, stream>>>(ogb, woutb, out, T_LEN, HID,
                                                 VAL_DIM);
  }
}
